// AttentionHead_17970143166900
// MI455X (gfx1250) — matrix operand order repaired and checked by reading, not compiled, not run on hardware
//
#include <hip/hip_runtime.h>
#include <hip/hip_bf16.h>

typedef __attribute__((ext_vector_type(16))) _Float16 v16h;
typedef __attribute__((ext_vector_type(8)))  _Float16 v8h;
typedef __attribute__((ext_vector_type(8)))  float    v8f;
typedef __attribute__((ext_vector_type(4)))  float    v4f;

#define TSEQ 2048
#define EMB  1024
#define NH   64
#define NB   4

// ---------------------------------------------------------------------------
// Kernel 0: bulk fp32 -> f16 conversion (8 elements per thread iteration).
// Run once for x and once per weight matrix so the GEMM hot loops carry no
// v_cvt work and x is read from HBM exactly once in fp32.
// ---------------------------------------------------------------------------
__global__ __launch_bounds__(256) void f32_to_f16_kernel(
    const float* __restrict__ src, _Float16* __restrict__ dst, int n8)
{
    int i = blockIdx.x * blockDim.x + threadIdx.x;
    const int stride = gridDim.x * blockDim.x;
    for (; i < n8; i += stride) {
        const v4f* s = (const v4f*)(src + (size_t)i * 8);
        v4f a = s[0], b = s[1];
        v8h o;
#pragma unroll
        for (int j = 0; j < 4; ++j) {
            o[j]     = (_Float16)a[j];
            o[4 + j] = (_Float16)b[j];
        }
        *(v8h*)(dst + (size_t)i * 8) = o;
    }
}

// ---------------------------------------------------------------------------
// Kernel 1: fused QKV projection, all-f16 operands.
// One wave per 16-token tile computes Q, K AND V for that tile:
// per k-step load the A fragment once (2x b128) and run 12 WMMAs
// (3 matrices x 4 n-tiles).  Weights are L2-resident f16.
// Q,K stored row-major f16 [b][t][64]; V stored transposed [b][h][t].
// ---------------------------------------------------------------------------
__global__ __launch_bounds__(256) void qkv_proj_kernel(
    const _Float16* __restrict__ xh, const _Float16* __restrict__ Wh,
    _Float16* __restrict__ Qh, _Float16* __restrict__ Kh,
    _Float16* __restrict__ VhT)
{
    const int lane = threadIdx.x & 31;
    const int wid  = blockIdx.x * 8 + (threadIdx.x >> 5);   // 512 tiles
    const int b    = wid >> 7;
    const int q0   = (wid & 127) << 4;
    const int l15  = lane & 15;
    const int hi   = (lane >> 4) & 1;

    const _Float16* __restrict__ xrow =
        xh + (size_t)(b * TSEQ + q0 + l15) * EMB;

    v8f acc[3][4];
#pragma unroll
    for (int m = 0; m < 3; ++m)
#pragma unroll
        for (int nt = 0; nt < 4; ++nt) acc[m][nt] = v8f{};

    for (int kb = 0; kb < EMB; kb += 32) {
        // A fragment: 16x32 f16, lane(M=l15) holds K = kb + hi*16 + g*8 + j
        v16h a;
#pragma unroll
        for (int g = 0; g < 2; ++g) {
            const v8h t = *(const v8h*)(xrow + kb + hi * 16 + g * 8);
#pragma unroll
            for (int j = 0; j < 8; ++j) a[g * 8 + j] = t[j];
        }
#pragma unroll
        for (int m = 0; m < 3; ++m) {
            const _Float16* __restrict__ Wm = Wh + (size_t)m * (NH * EMB);
#pragma unroll
            for (int nt = 0; nt < 4; ++nt) {
                // B[k,n] = W[n,k]: lane(N = nt*16+l15) reads 16 contig f16
                const _Float16* wp =
                    Wm + (size_t)(nt * 16 + l15) * EMB + kb + hi * 16;
                v8h w0 = *(const v8h*)wp;
                v8h w1 = *(const v8h*)(wp + 8);
                v16h bf;
#pragma unroll
                for (int j = 0; j < 8; ++j) { bf[j] = w0[j]; bf[8 + j] = w1[j]; }
                acc[m][nt] = __builtin_amdgcn_wmma_f32_16x16x32_f16(
                    false, a, false, bf, (short)0, acc[m][nt], false, false);
            }
        }
    }

    // C/D layout: lane -> N = l15 (+nt*16); VGPR g -> M = g + hi*8
#pragma unroll
    for (int nt = 0; nt < 4; ++nt)
#pragma unroll
        for (int g = 0; g < 8; ++g) {
            const int q = g + hi * 8;
            const int h = nt * 16 + l15;
            Qh [(size_t)(b * TSEQ + q0 + q) * NH + h]   = (_Float16)acc[0][nt][g];
            Kh [(size_t)(b * TSEQ + q0 + q) * NH + h]   = (_Float16)acc[1][nt][g];
            VhT[(size_t)(b * NH + h) * TSEQ + q0 + q]   = (_Float16)acc[2][nt][g];
        }
}

// ---------------------------------------------------------------------------
// Kernel 2: flash attention.  One wave per 16-query tile; stream keys in
// chunks of 32 with online softmax.  P is reshaped C/D-layout -> A-layout
// via a per-wave private LDS tile (same-wave DS ops are in-order).
// ---------------------------------------------------------------------------
__global__ __launch_bounds__(256) void attn_kernel(
    const _Float16* __restrict__ Qh, const _Float16* __restrict__ Kh,
    const _Float16* __restrict__ VhT, float* __restrict__ out)
{
    __shared__ __align__(16) _Float16 plds[8 * 512];   // 16x32 f16 per wave

    const int lane = threadIdx.x & 31;
    const int wib  = threadIdx.x >> 5;
    const int wid  = blockIdx.x * 8 + wib;
    const int b    = wid >> 7;
    const int q0   = (wid & 127) << 4;
    const int l15  = lane & 15;
    const int hi   = (lane >> 4) & 1;
    _Float16* __restrict__ myp = plds + wib * 512;

    // Q A-fragments for the two 32-deep K-steps over head dim (NH = 64)
    v16h aQ[2];
    const _Float16* __restrict__ qrow =
        Qh + (size_t)(b * TSEQ + q0 + l15) * NH;
#pragma unroll
    for (int s = 0; s < 2; ++s)
#pragma unroll
        for (int g = 0; g < 2; ++g) {
            const v8h t = *(const v8h*)(qrow + s * 32 + hi * 16 + g * 8);
#pragma unroll
            for (int j = 0; j < 8; ++j) aQ[s][g * 8 + j] = t[j];
        }

    float mrow[8], lrow[8];
    v8f O[4] = {v8f{}, v8f{}, v8f{}, v8f{}};
#pragma unroll
    for (int g = 0; g < 8; ++g) { mrow[g] = -3.0e38f; lrow[g] = 0.0f; }

    for (int k0 = 0; k0 <= q0 + 15; k0 += 32) {
        // ---- scores: two 16x16 tiles, contraction K=64 split in two steps
        v8f S[2];
#pragma unroll
        for (int t = 0; t < 2; ++t) {
            v8f s = {};
#pragma unroll
            for (int st = 0; st < 2; ++st) {
                // B[h, kk] = K[kk, h]: lane(N = key token) reads 16 contig f16
                const _Float16* kp =
                    Kh + (size_t)(b * TSEQ + k0 + t * 16 + l15) * NH +
                    st * 32 + hi * 16;
                v8h kv0 = *(const v8h*)kp;
                v8h kv1 = *(const v8h*)(kp + 8);
                v16h bK;
#pragma unroll
                for (int j = 0; j < 8; ++j) { bK[j] = kv0[j]; bK[8 + j] = kv1[j]; }
                s = __builtin_amdgcn_wmma_f32_16x16x32_f16(
                    false, aQ[st], false, bK, (short)0, s, false, false);
            }
#pragma unroll
            for (int g = 0; g < 8; ++g) S[t][g] = s[g] * 32.0f;  // * sqrt(c)
        }

        // ---- causal mask (only diagonal-straddling chunks need it)
        if (k0 + 31 > q0) {
#pragma unroll
            for (int t = 0; t < 2; ++t) {
                const int kk = k0 + t * 16 + l15;
#pragma unroll
                for (int g = 0; g < 8; ++g) {
                    const int q = q0 + g + hi * 8;
                    if (kk > q) S[t][g] = -3.0e38f;
                }
            }
        }

        // ---- online softmax update (rows live across the 16-lane N group)
#pragma unroll
        for (int g = 0; g < 8; ++g) {
            float r = fmaxf(S[0][g], S[1][g]);
            r = fmaxf(r, __shfl_xor(r, 1, 32));
            r = fmaxf(r, __shfl_xor(r, 2, 32));
            r = fmaxf(r, __shfl_xor(r, 4, 32));
            r = fmaxf(r, __shfl_xor(r, 8, 32));
            const float mn    = fmaxf(mrow[g], r);
            const float alpha = __expf(mrow[g] - mn);
            const float p0 = __expf(S[0][g] - mn);
            const float p1 = __expf(S[1][g] - mn);
            S[0][g] = p0; S[1][g] = p1;
            float rs = p0 + p1;
            rs += __shfl_xor(rs, 1, 32);
            rs += __shfl_xor(rs, 2, 32);
            rs += __shfl_xor(rs, 4, 32);
            rs += __shfl_xor(rs, 8, 32);
            lrow[g] = lrow[g] * alpha + rs;
            mrow[g] = mn;
            O[0][g] *= alpha; O[1][g] *= alpha;
            O[2][g] *= alpha; O[3][g] *= alpha;
        }

        // ---- P: C/D layout -> LDS (row-major 16x32 f16) -> A layout
#pragma unroll
        for (int t = 0; t < 2; ++t)
#pragma unroll
            for (int g = 0; g < 8; ++g)
                myp[(g + hi * 8) * 32 + t * 16 + l15] = (_Float16)S[t][g];

        asm volatile("s_wait_dscnt 0" ::: "memory");  // order store -> gather

        v16h aP;
#pragma unroll
        for (int g2 = 0; g2 < 2; ++g2) {
            const v8h t = *(const v8h*)(myp + l15 * 32 + hi * 16 + g2 * 8);
#pragma unroll
            for (int j = 0; j < 8; ++j) aP[g2 * 8 + j] = t[j];
        }

        // ---- O += P @ V   (B[kk, h] = VhT[h][kk], contiguous per lane)
#pragma unroll
        for (int ht = 0; ht < 4; ++ht) {
            const _Float16* vp =
                VhT + (size_t)(b * NH + ht * 16 + l15) * TSEQ + k0 + hi * 16;
            v8h vv0 = *(const v8h*)vp;
            v8h vv1 = *(const v8h*)(vp + 8);
            v16h bV;
#pragma unroll
            for (int j = 0; j < 8; ++j) { bV[j] = vv0[j]; bV[8 + j] = vv1[j]; }
            O[ht] = __builtin_amdgcn_wmma_f32_16x16x32_f16(
                false, aP, false, bV, (short)0, O[ht], false, false);
        }
    }

    // ---- normalize and store fp32 output [b][t][64]
#pragma unroll
    for (int ht = 0; ht < 4; ++ht) {
        const int h = ht * 16 + l15;
#pragma unroll
        for (int g = 0; g < 8; ++g) {
            const int q = g + hi * 8;
            out[(size_t)(b * TSEQ + q0 + q) * NH + h] = O[ht][g] / lrow[g];
        }
    }
}

extern "C" void kernel_launch(void* const* d_in, const int* in_sizes, int n_in,
                              void* d_out, int out_size, void* d_ws, size_t ws_size,
                              hipStream_t stream) {
    (void)in_sizes; (void)n_in; (void)out_size; (void)ws_size;
    const float* x  = (const float*)d_in[0];
    const float* Wk = (const float*)d_in[1];
    const float* Wq = (const float*)d_in[2];
    const float* Wv = (const float*)d_in[3];
    float* out = (float*)d_out;

    const size_t elems = (size_t)NB * TSEQ * NH;       // 524288 per tensor
    const size_t xsz   = (size_t)NB * TSEQ * EMB;      // 8388608
    const size_t wsz   = (size_t)NH * EMB;             // 65536 per matrix

    _Float16* Qh  = (_Float16*)d_ws;
    _Float16* Kh  = Qh  + elems;
    _Float16* VhT = Kh  + elems;
    _Float16* xh  = VhT + elems;
    _Float16* Wh  = xh  + xsz;          // [3][64][1024]: Q, K, V order

    // ---- pre-convert to f16 (x read from HBM exactly once in fp32)
    f32_to_f16_kernel<<<4096, 256, 0, stream>>>(x,  xh,            (int)(xsz / 8));
    f32_to_f16_kernel<<<  32, 256, 0, stream>>>(Wq, Wh,            (int)(wsz / 8));
    f32_to_f16_kernel<<<  32, 256, 0, stream>>>(Wk, Wh + wsz,      (int)(wsz / 8));
    f32_to_f16_kernel<<<  32, 256, 0, stream>>>(Wv, Wh + 2 * wsz,  (int)(wsz / 8));

    // ---- fused QKV projection: 512 tiles = 64 blocks of 8 waves
    qkv_proj_kernel<<<64, 256, 0, stream>>>(xh, Wh, Qh, Kh, VhT);

    // ---- flash attention: 512 query tiles = 64 blocks of 8 waves
    attn_kernel<<<64, 256, 0, stream>>>(Qh, Kh, VhT, out);
}
